// MultiHeadSelfAttention_60567628808552
// MI455X (gfx1250) — matrix operand order repaired and checked by reading, not compiled, not run on hardware
//
#include <hip/hip_runtime.h>
#include <stdint.h>

typedef __attribute__((ext_vector_type(16))) __bf16 v16bf;
typedef __attribute__((ext_vector_type(8)))  __bf16 v8bf;
typedef __attribute__((ext_vector_type(4)))  __bf16 v4bf;
typedef __attribute__((ext_vector_type(2)))  __bf16 v2bf;
typedef __attribute__((ext_vector_type(8)))  float  v8f;

constexpr int Bq = 4, Nq = 2048, Eq = 512, Hq = 8, DK = 64, DV = 64;
constexpr int QKVO = Hq * (2 * DK + DV);   // 1536
constexpr int ROWS = 2 * DK + DV;          // 192: per-(b,h) row stride in reinterpreted qkv

#define CAT16(lo, hi) __builtin_shufflevector((lo), (hi), 0,1,2,3,4,5,6,7,8,9,10,11,12,13,14,15)

__device__ __forceinline__ v8f wmma_bf16(v16bf a, v16bf b, v8f c) {
    return __builtin_amdgcn_wmma_f32_16x16x32_bf16(false, a, false, b, (short)0, c,
                                                   false, false);
}

// Flat shared pointer -> 32-bit LDS byte address (aperture maps addr[31:0] to LDS).
__device__ __forceinline__ unsigned lds_addr(const void* p) {
    return (unsigned)(uintptr_t)p;
}
// CDNA5 async memory->LDS copy (ASYNCcnt), VGLOBAL GV mode: VDST = LDS address.
__device__ __forceinline__ void async_ld_b128(unsigned lds, const void* gaddr) {
    asm volatile("global_load_async_to_lds_b128 %0, %1, off"
                 :: "v"(lds), "v"(gaddr) : "memory");
}
__device__ __forceinline__ void wait_async0() {
    asm volatile("s_wait_asynccnt 0x0" ::: "memory");
}

// Pure-VALU 16-lane butterfly reductions via DPP (no ds_bpermute round-trips).
template <int CTRL>
__device__ __forceinline__ float dpp_f(float x) {
    return __builtin_bit_cast(float,
        __builtin_amdgcn_update_dpp(0, __builtin_bit_cast(int, x), CTRL, 0xf, 0xf, true));
}
__device__ __forceinline__ float rmax16(float x) {
    x = fmaxf(x, dpp_f<0xB1>(x));   // quad_perm(1,0,3,2)
    x = fmaxf(x, dpp_f<0x4E>(x));   // quad_perm(2,3,0,1)
    x = fmaxf(x, dpp_f<0x141>(x));  // row_half_mirror
    x = fmaxf(x, dpp_f<0x140>(x));  // row_mirror
    return x;
}
__device__ __forceinline__ float rsum16(float x) {
    x += dpp_f<0xB1>(x);
    x += dpp_f<0x4E>(x);
    x += dpp_f<0x141>(x);
    x += dpp_f<0x140>(x);
    return x;
}

// Elementwise f32 -> bf16 (vec4), native hardware conversion.
__global__ __launch_bounds__(256) void cvt_f32_bf16(const float* __restrict__ in,
                                                    __bf16* __restrict__ out, int n4) {
    const int i = blockIdx.x * blockDim.x + threadIdx.x;
    if (i < n4) {
        const float4 f = ((const float4*)in)[i];
        v4bf o;
        o[0] = (__bf16)f.x; o[1] = (__bf16)f.y; o[2] = (__bf16)f.z; o[3] = (__bf16)f.w;
        ((v4bf*)out)[i] = o;
    }
}

// C[M,Nout] = A[M,K] @ Bm[K,Nout] + bias. A,Bm bf16; C f32 or bf16.
// Block = 8 waves sharing one 16-col strip (512x16 C): the 32x16 B tile is
// cooperatively transposed into LDS once per k-step and read back as one
// contiguous 32B fragment per lane. 4 WMMAs per wave per k-step.
// scale_q: multiply result by 8 for columns with col%192<64 (the q elements of
// the raw (B,N,1536)->(B,H,N,192) reshape) -- folds the reference's qk/DK**-0.5.
template <bool OUT_BF16>
__global__ __launch_bounds__(256) void gemm_bf16_wmma(
    const __bf16* __restrict__ A, const __bf16* __restrict__ Bm,
    const float* __restrict__ bias, float* __restrict__ Cf, __bf16* __restrict__ Cb,
    int M, int K, int Nout, int scale_q) {
    const int tid  = threadIdx.x;
    const int wave = tid >> 5;
    const int lane = tid & 31;
    const int hl   = lane >> 4;
    const int l16  = lane & 15;
    const int tilesN = Nout >> 4;
    const int TM = blockIdx.x / tilesN;          // 512-row super-strip
    const int tn = blockIdx.x % tilesN;
    const int row0 = TM * 512 + wave * 64;
    const int col  = tn * 16 + l16;

    __shared__ __attribute__((aligned(32))) __bf16 bt[16 * 32];   // [n][k] transposed B tile

    const int bk = tid >> 3;                     // staging: source row k (0..31)
    const int bn = (tid & 7) * 2;                // staging: source col pair (0..14)

    v8f acc[4] = {v8f{}, v8f{}, v8f{}, v8f{}};
    for (int kb = 0; kb < K; kb += 32) {
        // stage B tile transposed: bt[n][k] = Bm[kb+k][tn*16+n]
        const v2bf w = *(const v2bf*)(Bm + (size_t)(kb + bk) * Nout + tn * 16 + bn);
        bt[(bn + 0) * 32 + bk] = w[0];
        bt[(bn + 1) * 32 + bk] = w[1];
        __syncthreads();
        // B fragment: k = kb + 16*hl + e  -> contiguous 32B in bt
        const v16bf bf = *(const v16bf*)(bt + l16 * 32 + 16 * hl);
#pragma unroll
        for (int t = 0; t < 4; ++t) {
            const int row = row0 + t * 16 + l16;
            const __bf16* ap = A + (size_t)row * K + kb + hl * 16;
            acc[t] = wmma_bf16(CAT16(*(const v8bf*)ap, *(const v8bf*)(ap + 8)), bf, acc[t]);
        }
        __syncthreads();                         // WAR before next staging
    }
    const float sc = (scale_q && (col % ROWS) < DK) ? 8.0f : 1.0f;  // exact in bf16
    const float bv = bias[col];
#pragma unroll
    for (int t = 0; t < 4; ++t)
#pragma unroll
        for (int v = 0; v < 8; ++v) {
            const int m = row0 + t * 16 + v + 8 * hl;     // C/D: row = v + 8*half
            const float r = (acc[t][v] + bv) * sc;
            if (OUT_BF16) Cb[(size_t)m * Nout + col] = (__bf16)r;
            else          Cf[(size_t)m * Nout + col] = r;
        }
}

// Flash attention over bf16 qkv reinterpreted per (b,h): rows of 192 bf16
// (q at +0 pre-scaled by 8, k at +64, v at +128). Block = 8 waves x 16 query
// rows, all waves share one (b,h): K/V tiles staged in LDS per 32-key block
// (K via async DMA, V transposed).
__global__ __launch_bounds__(256) void attn_flash_wmma(
    const __bf16* __restrict__ qkv, __bf16* __restrict__ y) {
    const int tid  = threadIdx.x;
    const int wave = tid >> 5;
    const int lane = tid & 31;
    const int hl   = lane >> 4;
    const int l16  = lane & 15;
    const int bh = blockIdx.x >> 4;              // 16 blocks per (b,h)
    const int qb = (blockIdx.x & 15) * 8 + wave; // query block 0..127
    const __bf16* base = qkv + (size_t)bh * Nq * ROWS;

    __shared__ __attribute__((aligned(32))) __bf16 kbuf[32 * 64];      // [key][d]
    __shared__ __attribute__((aligned(32))) __bf16 vt[64 * 32];        // [dv][key]
    __shared__ __attribute__((aligned(32))) __bf16 pstage[8 * 16 * 32];
    __bf16* pw = pstage + wave * (16 * 32);

    // Q as two A-fragments (per-wave private rows, straight b128 loads)
    const int qrow = qb * 16 + l16;
    const __bf16* qp = base + (size_t)qrow * ROWS;
    v16bf qf[2];
#pragma unroll
    for (int s = 0; s < 2; ++s) {
        const __bf16* p = qp + s * 32 + hl * 16;
        qf[s] = CAT16(*(const v8bf*)p, *(const v8bf*)(p + 8));
    }

    // K staging: one async 16B transfer per thread covers the 4KB tile
    const int ak = tid >> 3;                     // key row 0..31
    const int ac = tid & 7;                      // 16B chunk in 128B row
    const unsigned koff = lds_addr(kbuf + ak * 64 + ac * 8);

    float mrow[8], lrow[8];
    v8f o[4] = {v8f{}, v8f{}, v8f{}, v8f{}};
#pragma unroll
    for (int v = 0; v < 8; ++v) { mrow[v] = -3.0e38f; lrow[v] = 0.0f; }

    for (int jb = 0; jb < Nq; jb += 32) {
        // K tile: memory->LDS async DMA (fragment order == memory order)
        async_ld_b128(koff, base + (size_t)(jb + ak) * ROWS + DK + ac * 8);
        // V tile: transpose-stage vt[dv][key]
#pragma unroll
        for (int c2 = 0; c2 < 2; ++c2) {
            const int c   = tid * 2 + c2;
            const int key = c >> 4;
            const int dvg = (c & 15) * 4;
            const v4bf w = *(const v4bf*)(base + (size_t)(jb + key) * ROWS + 2 * DK + dvg);
#pragma unroll
            for (int i = 0; i < 4; ++i) vt[(dvg + i) * 32 + key] = w[i];
        }
        wait_async0();
        __syncthreads();

        // S = Qs @ K^T (q pre-scaled by 8 in stage 1)
        v8f s0 = {}, s1 = {};
#pragma unroll
        for (int s = 0; s < 2; ++s) {
            const v16bf kf0 = *(const v16bf*)(kbuf + l16 * 64 + s * 32 + 16 * hl);
            const v16bf kf1 = *(const v16bf*)(kbuf + (16 + l16) * 64 + s * 32 + 16 * hl);
            s0 = wmma_bf16(qf[s], kf0, s0);
            s1 = wmma_bf16(qf[s], kf1, s1);
        }
        // Online softmax across 32 keys; row m = v + 8*hl lives in 16 lanes of one half.
        float p0e[8], p1e[8];
#pragma unroll
        for (int v = 0; v < 8; ++v) {
            const float a = s0[v], b = s1[v];
            const float mx = rmax16(fmaxf(a, b));          // DPP butterfly, no DS
            const float mn = fmaxf(mrow[v], mx);
            const float e0 = __expf(a - mn), e1 = __expf(b - mn);
            const float ls = rsum16(e0 + e1);              // DPP butterfly, no DS
            const float alpha = __expf(mrow[v] - mn);
            lrow[v] = lrow[v] * alpha + ls;
            mrow[v] = mn;
            o[0][v] *= alpha; o[1][v] *= alpha; o[2][v] *= alpha; o[3][v] *= alpha;
            p0e[v] = e0; p1e[v] = e1;
        }
        // Transpose P (C-layout -> A-layout) through wave-private LDS.
#pragma unroll
        for (int v = 0; v < 8; ++v) {
            const int m = v + 8 * hl;
            pw[m * 32 + l16]      = (__bf16)p0e[v];
            pw[m * 32 + 16 + l16] = (__bf16)p1e[v];
        }
        const __bf16* pr = pw + l16 * 32 + hl * 16;
        const v16bf pf = CAT16(*(const v8bf*)pr, *(const v8bf*)(pr + 8));
        // O += P @ V  (V fragment = contiguous 32B of transposed tile)
#pragma unroll
        for (int t = 0; t < 4; ++t) {
            const v16bf vf = *(const v16bf*)(vt + (t * 16 + l16) * 32 + 16 * hl);
            o[t] = wmma_bf16(pf, vf, o[t]);
        }
        __syncthreads();                          // WAR before next staging
    }
    // Finalize: y[bh][m][dv] flat bf16 (= reference's raw reshape order)
#pragma unroll
    for (int v = 0; v < 8; ++v) {
        const int m = qb * 16 + v + 8 * hl;
        const float inv = 1.0f / lrow[v];
        const size_t rb = (size_t)bh * Nq * DV + (size_t)m * DV;
#pragma unroll
        for (int t = 0; t < 4; ++t) y[rb + t * 16 + l16] = (__bf16)(o[t][v] * inv);
    }
}

extern "C" void kernel_launch(void* const* d_in, const int* in_sizes, int n_in,
                              void* d_out, int out_size, void* d_ws, size_t ws_size,
                              hipStream_t stream) {
    (void)in_sizes; (void)n_in; (void)out_size; (void)ws_size;
    const float* x     = (const float*)d_in[0];
    const float* W_qkv = (const float*)d_in[1];
    const float* b_qkv = (const float*)d_in[2];
    const float* W_o   = (const float*)d_in[3];
    const float* b_o   = (const float*)d_in[4];
    float* out = (float*)d_out;

    const int M = Bq * Nq;                              // 8192
    const size_t n_x    = (size_t)M * Eq;               // 4,194,304
    const size_t n_wqkv = (size_t)Eq * QKVO;            //   786,432
    const size_t n_wo   = (size_t)Hq * DV * Eq;         //   262,144
    const size_t n_qkv  = (size_t)M * QKVO;             // 12,582,912

    __bf16* x_bf    = (__bf16*)d_ws;
    __bf16* wqkv_bf = x_bf + n_x;
    __bf16* wo_bf   = wqkv_bf + n_wqkv;
    __bf16* qkv_bf  = wo_bf + n_wo;
    __bf16* y_bf    = qkv_bf + n_qkv;

    // Stage 0: one-time f32 -> bf16 conversion of inputs
    cvt_f32_bf16<<<(int)(n_x / 4 / 256),    256, 0, stream>>>(x,     x_bf,    (int)(n_x / 4));
    cvt_f32_bf16<<<(int)(n_wqkv / 4 / 256), 256, 0, stream>>>(W_qkv, wqkv_bf, (int)(n_wqkv / 4));
    cvt_f32_bf16<<<(int)(n_wo / 4 / 256),   256, 0, stream>>>(W_o,   wo_bf,   (int)(n_wo / 4));

    // Stage 1: qkv = x @ W_qkv + b_qkv, q pre-scaled by 8 (bf16 out)
    gemm_bf16_wmma<true><<<(M / 512) * (QKVO / 16), 256, 0, stream>>>(
        x_bf, wqkv_bf, b_qkv, nullptr, qkv_bf, M, Eq, QKVO, 1);

    // Stage 2: flash attention, 16 blocks per (b,h)
    attn_flash_wmma<<<Bq * Hq * 16, 256, 0, stream>>>(qkv_bf, y_bf);

    // Stage 3: out = y @ W_o + b_o  (f32 out)
    gemm_bf16_wmma<false><<<(M / 512) * (Eq / 16), 256, 0, stream>>>(
        y_bf, wo_bf, b_o, out, nullptr, M, Hq * DV, Eq, 0);
}
